// NonLocalBlock_82583631167586
// MI455X (gfx1250) — matrix operand order repaired and checked by reading, not compiled, not run on hardware
//
#include <hip/hip_runtime.h>
#include <hip/hip_bf16.h>

// ---------------------------------------------------------------------------
// NonLocal block for MI455X (gfx1250, wave32, WMMA bf16 path).
// All matmuls run through v_wmma_f32_16x16x32_bf16 with fp32 accumulation.
// Layouts chosen so every WMMA fragment is a contiguous 16B global load.
// 32x64 output tile per wave: 8 WMMAs per 12 b128 loads per K-step.
// x transpose uses GLOBAL_LOAD_ASYNC_TO_LDS_B128 (ASYNCcnt) staging.
// ---------------------------------------------------------------------------

typedef __bf16 bf16_t;
typedef __attribute__((ext_vector_type(16))) __bf16 v16bf;
typedef __attribute__((ext_vector_type(8)))  __bf16 v8bf;
typedef __attribute__((ext_vector_type(8)))  float  v8f;

#define BB   32
#define CC   512
#define CI   256
#define NN   3136          // 56*56
#define KPROJ 512

__device__ __forceinline__ bf16_t f2bf(float f) { return (bf16_t)f; }

// A fragment: 16x32 bf16, row-major A[M][K]; lane (m = lane&15, h = lane>>4)
// element e -> k = (e<8 ? 0 : 16) + h*8 + (e&7)  => two contiguous 8-elem runs.
__device__ __forceinline__ v16bf load_frag_a(const bf16_t* __restrict__ A,
                                             int lda, int lane) {
  const int m = lane & 15, h = lane >> 4;
  const bf16_t* p = A + (long)m * lda + h * 8;
  v8bf lo = *(const v8bf*)(p);
  v8bf hi = *(const v8bf*)(p + 16);
  v16bf r;
#pragma unroll
  for (int e = 0; e < 8; ++e) { r[e] = lo[e]; r[e + 8] = hi[e]; }
  return r;
}

// B fragment: 32x16 bf16, stored transposed BT[N][K]; lane (n = lane&15,
// h = lane>>4) element e -> k = (e<8 ? 0 : 16) + h*8 + (e&7)  => two contiguous 8-elem runs.
__device__ __forceinline__ v16bf load_frag_bt(const bf16_t* __restrict__ BT,
                                              int ldb, int lane) {
  const int n = lane & 15, h = lane >> 4;
  const bf16_t* p = BT + (long)n * ldb + h * 8;
  v8bf lo = *(const v8bf*)(p);
  v8bf hi = *(const v8bf*)(p + 16);
  v16bf r;
#pragma unroll
  for (int e = 0; e < 8; ++e) { r[e] = lo[e]; r[e + 8] = hi[e]; }
  return r;
}

__device__ __forceinline__ v8f wmma_bf16(v16bf a, v16bf b, v8f c) {
  return __builtin_amdgcn_wmma_f32_16x16x32_bf16(
      false, a, false, b, (short)0, c, false, false);
}

// 32(M)x64(N) tile GEMM inner loop; Ab points at (m0,k=0), Bb at (n0,k=0).
// PF: per-lane global_prefetch of the fragments 8 K-steps ahead.
template <int LDA, int LDB, int K, bool PF>
__device__ __forceinline__ void gemm32x64(const bf16_t* __restrict__ Ab,
                                          const bf16_t* __restrict__ Bb,
                                          v8f acc[2][4], int lane) {
  const int ml = lane & 15, h = lane >> 4;
  for (int k = 0; k < K; k += 32) {
    if (PF && (k + 256) < K) {
      __builtin_prefetch(Ab + (long)ml * LDA + h * 8 + k + 256);
      __builtin_prefetch(Ab + (long)(16 + ml) * LDA + h * 8 + k + 256);
#pragma unroll
      for (int j = 0; j < 4; ++j)
        __builtin_prefetch(Bb + (long)(j * 16 + ml) * LDB + h * 16 + k + 256);
    }
    v16bf a0 = load_frag_a(Ab + k, LDA, lane);
    v16bf a1 = load_frag_a(Ab + (long)16 * LDA + k, LDA, lane);
#pragma unroll
    for (int j = 0; j < 4; ++j) {
      v16bf bf = load_frag_bt(Bb + (long)j * 16 * LDB + k, LDB, lane);
      acc[0][j] = wmma_bf16(a0, bf, acc[0][j]);
      acc[1][j] = wmma_bf16(a1, bf, acc[1][j]);
    }
  }
}

// ---------------------------------------------------------------------------
// Prep kernels
// ---------------------------------------------------------------------------

__global__ __launch_bounds__(256) void k_zero(float* p, int n) {
  int i = blockIdx.x * 256 + threadIdx.x;
  if (i < n) p[i] = 0.f;
}

// concat+convert g_w/theta_w/phi_w -> wcat bf16 [768][512]
__global__ __launch_bounds__(256) void k_wcat(const float* __restrict__ g_w,
                                              const float* __restrict__ t_w,
                                              const float* __restrict__ p_w,
                                              bf16_t* __restrict__ wcat) {
  int i = blockIdx.x * 256 + threadIdx.x;          // 0 .. 768*512-1
  if (i >= 768 * KPROJ) return;
  int m = i >> 9;
  float v = (m < 256) ? g_w[i] : (m < 512) ? t_w[i - 256 * KPROJ]
                                           : p_w[i - 512 * KPROJ];
  wcat[i] = f2bf(v);
}

__global__ __launch_bounds__(256) void k_wb(const float* __restrict__ W_w,
                                            bf16_t* __restrict__ Wb16) {
  int i = blockIdx.x * 256 + threadIdx.x;          // 512*256
  if (i < CC * CI) Wb16[i] = f2bf(W_w[i]);
}

__global__ __launch_bounds__(256) void k_bcat(const float* __restrict__ g_b,
                                              const float* __restrict__ t_b,
                                              const float* __restrict__ p_b,
                                              float* __restrict__ bcat) {
  int i = blockIdx.x * 256 + threadIdx.x;
  if (i >= 768) return;
  bcat[i] = (i < 256) ? g_b[i] : (i < 512) ? t_b[i - 256] : p_b[i - 512];
}

// transpose+convert x[b][c][n] f32 -> xT[b][n][c] bf16.
// 32x32 f32 tile staged global->LDS with GLOBAL_LOAD_ASYNC_TO_LDS_B128
// (one float4 per lane, ASYNCcnt-tracked), then read/convert/store.
__global__ __launch_bounds__(256) void k_xt(const float* __restrict__ x,
                                            bf16_t* __restrict__ xT) {
  __shared__ float s[32][36];                      // 144B row stride, 16B aligned
  const int b = blockIdx.z, c0 = blockIdx.y * 32, n0 = blockIdx.x * 32;
  {
    const int t = threadIdx.x;
    const int c = t >> 3;                          // 0..31
    const int n4 = (t & 7) * 4;                    // 0,4,...,28
    const float* g = x + ((long)b * CC + c0 + c) * NN + n0 + n4;
    unsigned lds = (unsigned)(size_t)&s[c][n4];    // low 32 bits = LDS offset
    asm volatile("global_load_async_to_lds_b128 %0, %1, off"
                 :: "v"(lds), "v"(g) : "memory");
    asm volatile("s_wait_asynccnt 0" ::: "memory");
  }
  __syncthreads();
  bf16_t* dst = xT + ((long)b * NN + n0) * CC + c0;
#pragma unroll
  for (int i = 0; i < 4; ++i) {
    int lin = threadIdx.x + i * 256;
    int n = lin >> 5, c = lin & 31;
    dst[(long)n * CC + c] = f2bf(s[c][n]);
  }
}

// ---------------------------------------------------------------------------
// GEMM 1: projections.  out[m][n] = sum_c wcat[m][c]*x[b][c][n] + bcat[m]
// grid (49 n-tiles of 64, 3 proj of 256 rows, 32 batch); wave = 32x64 tile.
//   blockIdx.y == 0 -> g (stored transposed gxT[b][n][c'])
//   blockIdx.y == 1 -> theta [c'][N], == 2 -> phi [c'][N]
// ---------------------------------------------------------------------------
__global__ __launch_bounds__(256) void k_proj(const bf16_t* __restrict__ wcat,
                                              const float* __restrict__ bcat,
                                              const bf16_t* __restrict__ xT,
                                              bf16_t* __restrict__ gxT,
                                              bf16_t* __restrict__ th,
                                              bf16_t* __restrict__ ph) {
  const int lane = threadIdx.x & 31, wid = threadIdx.x >> 5;
  const int b = blockIdx.z;
  const int m0 = blockIdx.y * 256 + wid * 32;
  const long n0 = (long)blockIdx.x * 64;

  const bf16_t* Ab = wcat + (long)m0 * KPROJ;
  const bf16_t* Bb = xT + ((long)b * NN + n0) * KPROJ;

  v8f acc[2][4] = {};
  gemm32x64<KPROJ, KPROJ, KPROJ, false>(Ab, Bb, acc, lane);

  const int h = lane >> 4, col = lane & 15;
  if (blockIdx.y == 0) {                   // g: transposed, packed 16B stores
#pragma unroll
    for (int mi = 0; mi < 2; ++mi) {
      const int rb = m0 + mi * 16 + h * 8;
#pragma unroll
      for (int j = 0; j < 4; ++j) {
        long n = n0 + j * 16 + col;
        v8bf pk;
#pragma unroll
        for (int r = 0; r < 8; ++r) pk[r] = f2bf(acc[mi][j][r] + bcat[rb + r]);
        *(v8bf*)(gxT + ((long)b * NN + n) * CI + rb) = pk;
      }
    }
  } else {                                 // theta / phi: row-major [c'][N]
    bf16_t* base = (blockIdx.y == 1) ? th : ph;
    const int mloc = m0 - (int)blockIdx.y * 256;   // 0..255 within projection
#pragma unroll
    for (int mi = 0; mi < 2; ++mi) {
      const int rb = mloc + mi * 16 + h * 8;
      bf16_t* dst = base + ((long)b * CI + rb) * NN;
      const float* bc = bcat + (int)blockIdx.y * 256 + rb;
#pragma unroll
      for (int j = 0; j < 4; ++j) {
        long n = n0 + j * 16 + col;
#pragma unroll
        for (int r = 0; r < 8; ++r)
          dst[(long)r * NN + n] = f2bf(acc[mi][j][r] + bc[r]);
      }
    }
  }
}

// ---------------------------------------------------------------------------
// GEMM 2: Gram  f[b][i][j] = (1/N) * sum_n th[i][n]*ph[j][n]
// A = th [256][N] row-major; BT = ph [256][N].  grid (4 j-tiles of 64, 1, 32);
// 8 waves cover all 256 i rows.  Long-K streaming: prefetch enabled.
// ---------------------------------------------------------------------------
__global__ __launch_bounds__(256) void k_gram(const bf16_t* __restrict__ th,
                                              const bf16_t* __restrict__ ph,
                                              bf16_t* __restrict__ f16) {
  const int lane = threadIdx.x & 31, wid = threadIdx.x >> 5;
  const int b = blockIdx.z;
  const int m0 = wid * 32;
  const int j0 = blockIdx.x * 64;

  const bf16_t* Ab = th + ((long)b * CI + m0) * NN;
  const bf16_t* Bb = ph + ((long)b * CI + j0) * NN;

  v8f acc[2][4] = {};
  gemm32x64<NN, NN, NN, true>(Ab, Bb, acc, lane);

  const float inv = 1.0f / (float)NN;
  const int h = lane >> 4, col = lane & 15;
  bf16_t* dst = f16 + (long)b * CI * CI;
#pragma unroll
  for (int mi = 0; mi < 2; ++mi) {
    const int rb = m0 + mi * 16 + h * 8;
#pragma unroll
    for (int j = 0; j < 4; ++j)
#pragma unroll
      for (int r = 0; r < 8; ++r)
        dst[(long)(rb + r) * CI + j0 + j * 16 + col] = f2bf(acc[mi][j][r] * inv);
  }
}

// ---------------------------------------------------------------------------
// GEMM 3: y[b][i][n] = sum_j f[i][j]*g[j][n];  stored transposed yT[b][n][i].
// A = f16 [256][256]; BT = gxT[b] [N][256].  grid (49, 1, 32).
// ---------------------------------------------------------------------------
__global__ __launch_bounds__(256) void k_y(const bf16_t* __restrict__ f16,
                                           const bf16_t* __restrict__ gxT,
                                           bf16_t* __restrict__ yT) {
  const int lane = threadIdx.x & 31, wid = threadIdx.x >> 5;
  const int b = blockIdx.z;
  const int m0 = wid * 32;
  const long n0 = (long)blockIdx.x * 64;

  const bf16_t* Ab = f16 + (long)b * CI * CI + (long)m0 * CI;
  const bf16_t* Bb = gxT + ((long)b * NN + n0) * CI;

  v8f acc[2][4] = {};
  gemm32x64<CI, CI, CI, false>(Ab, Bb, acc, lane);

  const int h = lane >> 4, col = lane & 15;
#pragma unroll
  for (int mi = 0; mi < 2; ++mi) {
    const int rb = m0 + mi * 16 + h * 8;
#pragma unroll
    for (int j = 0; j < 4; ++j) {
      long n = n0 + j * 16 + col;
      v8bf pk;
#pragma unroll
      for (int r = 0; r < 8; ++r) pk[r] = f2bf(acc[mi][j][r]);
      *(v8bf*)(yT + ((long)b * NN + n) * CI + rb) = pk;
    }
  }
}

// ---------------------------------------------------------------------------
// GEMM 4: z[b][o][n] = sum_j W[o][j]*y[j][n] + W_b[o] + x[b][o][n]
// writes z to d_out (fp32) and accumulates per-channel sum / sumsq for BN.
// A = Wb16 [512][256]; BT = yT[b] [N][256].  grid (49, 2, 32).
// ---------------------------------------------------------------------------
__global__ __launch_bounds__(256) void k_out(const bf16_t* __restrict__ Wb16,
                                             const float* __restrict__ W_b,
                                             const bf16_t* __restrict__ yT,
                                             const float* __restrict__ x,
                                             float* __restrict__ z,
                                             float* __restrict__ sums,
                                             float* __restrict__ sumsq) {
  const int lane = threadIdx.x & 31, wid = threadIdx.x >> 5;
  const int b = blockIdx.z;
  const int m0 = blockIdx.y * 256 + wid * 32;
  const long n0 = (long)blockIdx.x * 64;

  const bf16_t* Ab = Wb16 + (long)m0 * CI;
  const bf16_t* Bb = yT + ((long)b * NN + n0) * CI;

  v8f acc[2][4] = {};
  gemm32x64<CI, CI, CI, false>(Ab, Bb, acc, lane);

  const int h = lane >> 4, col = lane & 15;
#pragma unroll
  for (int mi = 0; mi < 2; ++mi) {
    const int rb = m0 + mi * 16 + h * 8;
    float s1[8], s2[8];
#pragma unroll
    for (int r = 0; r < 8; ++r) { s1[r] = 0.f; s2[r] = 0.f; }
#pragma unroll
    for (int r = 0; r < 8; ++r) {
      const int o = rb + r;
      const float wb = W_b[o];
      const float* xr = x + ((long)b * CC + o) * NN;
      float* zr = z + ((long)b * CC + o) * NN;
#pragma unroll
      for (int j = 0; j < 4; ++j) {
        long n = n0 + j * 16 + col;
        float v = acc[mi][j][r] + wb + xr[n];
        zr[n] = v;
        s1[r] += v;
        s2[r] += v * v;
      }
    }
    // reduce over the 16 lanes sharing the same rows (xor 1..8 stays in-half)
#pragma unroll
    for (int r = 0; r < 8; ++r) {
#pragma unroll
      for (int off = 8; off >= 1; off >>= 1) {
        s1[r] += __shfl_xor(s1[r], off, 32);
        s2[r] += __shfl_xor(s2[r], off, 32);
      }
    }
    if (col == 0) {
#pragma unroll
      for (int r = 0; r < 8; ++r) {
        atomicAdd(&sums[rb + r], s1[r]);
        atomicAdd(&sumsq[rb + r], s2[r]);
      }
    }
  }
}

// ---------------------------------------------------------------------------
// BN finalize
// ---------------------------------------------------------------------------
__global__ __launch_bounds__(256) void k_bnp(const float* __restrict__ sums,
                                             const float* __restrict__ sumsq,
                                             const float* __restrict__ gamma,
                                             const float* __restrict__ beta,
                                             float* __restrict__ scale,
                                             float* __restrict__ shift) {
  int c = blockIdx.x * 256 + threadIdx.x;
  if (c >= CC) return;
  const float cnt = (float)BB * (float)NN;
  float mean = sums[c] / cnt;
  float var = sumsq[c] / cnt - mean * mean;
  float sc = gamma[c] * rsqrtf(var + 1e-5f);
  scale[c] = sc;
  shift[c] = beta[c] - mean * sc;
}

__global__ __launch_bounds__(256) void k_bn(float* __restrict__ z,
                                            const float* __restrict__ scale,
                                            const float* __restrict__ shift) {
  long i4 = (long)blockIdx.x * 256 + threadIdx.x;   // float4 index, N%4==0
  int c = (int)((i4 / (NN / 4)) & (CC - 1));
  float4 v = ((float4*)z)[i4];
  float sc = scale[c], sh = shift[c];
  v.x = v.x * sc + sh;
  v.y = v.y * sc + sh;
  v.z = v.z * sc + sh;
  v.w = v.w * sc + sh;
  ((float4*)z)[i4] = v;
}

// ---------------------------------------------------------------------------
// Host launcher
// ---------------------------------------------------------------------------
extern "C" void kernel_launch(void* const* d_in, const int* in_sizes, int n_in,
                              void* d_out, int out_size, void* d_ws, size_t ws_size,
                              hipStream_t stream) {
  const float* x    = (const float*)d_in[0];
  const float* g_w  = (const float*)d_in[1];
  const float* g_b  = (const float*)d_in[2];
  const float* t_w  = (const float*)d_in[3];
  const float* t_b  = (const float*)d_in[4];
  const float* p_w  = (const float*)d_in[5];
  const float* p_b  = (const float*)d_in[6];
  const float* W_w  = (const float*)d_in[7];
  const float* W_b  = (const float*)d_in[8];
  const float* gamma= (const float*)d_in[9];
  const float* beta = (const float*)d_in[10];
  float* z = (float*)d_out;

  char* ws = (char*)d_ws;
  size_t off = 0;
  bf16_t* xT   = (bf16_t*)(ws + off); off += (size_t)BB * NN * CC * 2;     // 102,760,448
  bf16_t* wcat = (bf16_t*)(ws + off); off += (size_t)768 * KPROJ * 2;      // 786,432
  float*  bcat = (float*) (ws + off); off += 768 * 4;                      // 3,072
  bf16_t* Wb16 = (bf16_t*)(ws + off); off += (size_t)CC * CI * 2;          // 262,144
  bf16_t* th   = (bf16_t*)(ws + off); off += (size_t)BB * CI * NN * 2;     // 51,380,224
  bf16_t* ph   = (bf16_t*)(ws + off); off += (size_t)BB * CI * NN * 2;
  bf16_t* gxT  = (bf16_t*)(ws + off); off += (size_t)BB * CI * NN * 2;
  bf16_t* f16  = (bf16_t*)(ws + off); off += (size_t)BB * CI * CI * 2;     // 4,194,304
  bf16_t* yT   = (bf16_t*)(ws + off); off += (size_t)BB * CI * NN * 2;
  float* sums  = (float*)(ws + off);  off += 512 * 4;
  float* sumsq = (float*)(ws + off);  off += 512 * 4;
  float* scale = (float*)(ws + off);  off += 512 * 4;
  float* shift = (float*)(ws + off);  off += 512 * 4;

  // prep
  k_zero<<<4, 256, 0, stream>>>(sums, 1024);                 // sums+sumsq contiguous
  k_wcat<<<(768 * KPROJ + 255) / 256, 256, 0, stream>>>(g_w, t_w, p_w, wcat);
  k_wb  <<<(CC * CI + 255) / 256, 256, 0, stream>>>(W_w, Wb16);
  k_bcat<<<3, 256, 0, stream>>>(g_b, t_b, p_b, bcat);
  k_xt  <<<dim3(NN / 32, CC / 32, BB), 256, 0, stream>>>(x, xT);

  // pipeline
  k_proj<<<dim3(NN / 64, 3, BB), 256, 0, stream>>>(wcat, bcat, xT, gxT, th, ph);
  k_gram<<<dim3(4, 1, BB), 256, 0, stream>>>(th, ph, f16);
  k_y   <<<dim3(NN / 64, 1, BB), 256, 0, stream>>>(f16, gxT, yT);
  k_out <<<dim3(NN / 64, 2, BB), 256, 0, stream>>>(Wb16, W_b, yT, x, z, sums, sumsq);
  k_bnp <<<2, 256, 0, stream>>>(sums, sumsq, gamma, beta, scale, shift);
  k_bn  <<<(size_t)BB * CC * NN / 4 / 256, 256, 0, stream>>>(z, scale, shift);
}
